// Masked_Attention_644245094967
// MI455X (gfx1250) — matrix operand order repaired and checked by reading, not compiled, not run on hardware
//
#include <hip/hip_runtime.h>
#include <hip/hip_bf16.h>

typedef __attribute__((ext_vector_type(16))) __bf16 v16bf;
typedef __attribute__((ext_vector_type(8)))  __bf16 v8bf;
typedef __attribute__((ext_vector_type(4)))  __bf16 v4bf;
typedef __attribute__((ext_vector_type(8)))  float  v8f;

#define WMMA_BF16(A, B, C) \
  __builtin_amdgcn_wmma_f32_16x16x32_bf16(false, (A), false, (B), (short)0, (C), false, false)

// A-fragment (16x32 bf16): lane = row; halves[0..15] = K half*16+j.
// p must already point at rowbase + k0 + half*16.
__device__ __forceinline__ v16bf ldA(const __bf16* __restrict__ p)
{
    v8bf x0 = *(const v8bf*)p;
    v8bf x1 = *(const v8bf*)(p + 8);
    v16bf a;
#pragma unroll
    for (int j = 0; j < 8; ++j) { a[j] = x0[j]; a[8 + j] = x1[j]; }
    return a;
}

// ---------------------------------------------------------------------------
// Kernel 0: f32 -> bf16 elementwise convert (vector-4).
// ---------------------------------------------------------------------------
__global__ __launch_bounds__(256)
void cvt_bf16_kernel(const float* __restrict__ in, __bf16* __restrict__ out, int n4)
{
    const int i = blockIdx.x * 256 + threadIdx.x;
    if (i < n4) {
        const float4 f = ((const float4*)in)[i];
        v4bf o;
        o[0] = (__bf16)f.x; o[1] = (__bf16)f.y; o[2] = (__bf16)f.z; o[3] = (__bf16)f.w;
        ((v4bf*)out)[i] = o;
    }
}

// ---------------------------------------------------------------------------
// Kernel 1: QKV projection (all-bf16 operands, f32 accum).
// xb[4096,768] @ wb[2304,768]^T -> Q(/8), K as [B*H][2048][64]; V^T [B*H][64][2048].
// Block = 256 threads (8 waves); wave computes 32x64; double-buffered K loop.
// ---------------------------------------------------------------------------
__global__ __launch_bounds__(256)
void qkv_gemm_kernel(const __bf16* __restrict__ x, const __bf16* __restrict__ w,
                     __bf16* __restrict__ Qo, __bf16* __restrict__ Ko,
                     __bf16* __restrict__ Vt)
{
    const int lane = threadIdx.x & 31;
    const int wave = threadIdx.x >> 5;
    const int half = lane >> 4;
    const int l16  = lane & 15;
    const int m0   = blockIdx.y * 256 + wave * 32;
    const int n0   = blockIdx.x * 64;

    const __bf16* a0p = x + (size_t)(m0 + l16) * 768 + half * 16;       // row group 0
    const __bf16* a1p = a0p + (size_t)16 * 768;                          // row group 1
    const __bf16* wp  = w + (size_t)(n0 + l16) * 768 + half * 16;        // + nb*16*768 + k0

    v8f acc[2][4] = {};
    v16bf a[2], b[4];
    a[0] = ldA(a0p); a[1] = ldA(a1p);
#pragma unroll
    for (int nb = 0; nb < 4; ++nb) b[nb] = *(const v16bf*)(wp + (size_t)nb * 16 * 768);

#pragma unroll 2
    for (int k0 = 0; k0 < 768; k0 += 32) {
        const int kn = (k0 + 32 < 768) ? k0 + 32 : 0;
        // prefetch next-step fragments (stay in flight across the WMMAs below)
        v16bf an[2], bn[4];
        an[0] = ldA(a0p + kn); an[1] = ldA(a1p + kn);
#pragma unroll
        for (int nb = 0; nb < 4; ++nb) bn[nb] = *(const v16bf*)(wp + (size_t)nb * 16 * 768 + kn);
        // 8 WMMAs on the resident fragments
#pragma unroll
        for (int rg = 0; rg < 2; ++rg)
#pragma unroll
            for (int nb = 0; nb < 4; ++nb)
                acc[rg][nb] = WMMA_BF16(a[rg], b[nb], acc[rg][nb]);
        // rotate (renamed away by unroll-2)
        a[0] = an[0]; a[1] = an[1];
#pragma unroll
        for (int nb = 0; nb < 4; ++nb) b[nb] = bn[nb];
    }

    // Epilogue: scatter into Q / K / V^T
#pragma unroll
    for (int nb = 0; nb < 4; ++nb) {
        const int col   = n0 + nb * 16 + l16;   // 0..2303
        const int which = col / 768;            // 0=Q 1=K 2=V
        const int rem   = col % 768;
        const int h     = rem >> 6;
        const int d     = rem & 63;
#pragma unroll
        for (int rg = 0; rg < 2; ++rg)
#pragma unroll
            for (int g = 0; g < 8; ++g) {
                const int m   = m0 + rg * 16 + g + 8 * half;
                const int bi  = m >> 11;
                const int tok = m & 2047;
                const int bh  = bi * 12 + h;
                const float v = acc[rg][nb][g];
                if (which == 0)
                    Qo[((size_t)bh * 2048 + tok) * 64 + d] = (__bf16)(v * 0.125f);
                else if (which == 1)
                    Ko[((size_t)bh * 2048 + tok) * 64 + d] = (__bf16)v;
                else
                    Vt[((size_t)bh * 64 + d) * 2048 + tok] = (__bf16)v;
            }
    }
}

// ---------------------------------------------------------------------------
// Kernel 2: flash attention, one wave per 16-query tile per (b,h).
// Pipelined: S-WMMAs -> issue next K-slab + current V loads -> softmax VALU
// (overlaps loads) -> per-wave LDS transpose (no block barrier) -> PV WMMAs.
// ---------------------------------------------------------------------------
__global__ __launch_bounds__(128)
void flash_attn_kernel(const __bf16* __restrict__ Q, const __bf16* __restrict__ K,
                       const __bf16* __restrict__ Vt, const float* __restrict__ mask,
                       __bf16* __restrict__ out)
{
    __shared__ __bf16 ptile[4][16 * 32];

    const int lane = threadIdx.x & 31;
    const int wave = threadIdx.x >> 5;
    const int half = lane >> 4;
    const int l16  = lane & 15;

    const int wg = blockIdx.x * 4 + wave;
    const int bh = wg >> 7;
    const int qt = wg & 127;
    const int bi = bh / 12;
    const int h  = bh % 12;
    const int q0 = qt * 16;

    // Q A-fragments (d = 0..31, 32..63), loaded once
    v16bf qa[2];
    {
        const __bf16* qrow = Q + ((size_t)bh * 2048 + q0 + l16) * 64;
        qa[0] = ldA(qrow + half * 16);
        qa[1] = ldA(qrow + 32 + half * 16);
    }

    float m_i[8], l_i[8];
    v8f   acc[4] = {};
#pragma unroll
    for (int g = 0; g < 8; ++g) { m_i[g] = -3.0e38f; l_i[g] = 0.0f; }

    const float*  mrow  = mask + (size_t)bi * 2048;
    const __bf16* kbase = K + (size_t)bh * 2048 * 64;
    const __bf16* vbase = Vt + (size_t)bh * 64 * 2048;
    __bf16* pt = ptile[wave];

    // preload K slab kk=0
    v16bf kb0[2], kb1[2];
    float mb[2];
#pragma unroll
    for (int nb = 0; nb < 2; ++nb) {
        const int key = nb * 16 + l16;
        const __bf16* krow = kbase + (size_t)key * 64 + half * 16;
        kb0[nb] = *(const v16bf*)krow;
        kb1[nb] = *(const v16bf*)(krow + 32);
        mb[nb]  = mrow[key] * -1000.0f;
    }

#pragma unroll 2
    for (int kk = 0; kk < 2048; kk += 32) {
        // ---- S = Q K^T + mask bias (resident fragments) ----
        v8f s[2];
#pragma unroll
        for (int nb = 0; nb < 2; ++nb) {
            v8f c0 = {};
            c0 = WMMA_BF16(qa[0], kb0[nb], c0);
            c0 = WMMA_BF16(qa[1], kb1[nb], c0);
#pragma unroll
            for (int g = 0; g < 8; ++g) s[nb][g] = c0[g] + mb[nb];
        }

        // ---- issue next K slab + current V fragments (in flight over softmax) ----
        const int kn = (kk + 32 < 2048) ? kk + 32 : 0;
        v16bf kn0[2], kn1[2];
        float mbn[2];
#pragma unroll
        for (int nb = 0; nb < 2; ++nb) {
            const int key = kn + nb * 16 + l16;
            const __bf16* krow = kbase + (size_t)key * 64 + half * 16;
            kn0[nb] = *(const v16bf*)krow;
            kn1[nb] = *(const v16bf*)(krow + 32);
            mbn[nb] = mrow[key] * -1000.0f;
        }
        v16bf bv[4];
#pragma unroll
        for (int nb = 0; nb < 4; ++nb)
            bv[nb] = *(const v16bf*)(vbase + (size_t)(nb * 16 + l16) * 2048 + kk + half * 16);

        // ---- online softmax (VALU; overlaps the loads above) ----
#pragma unroll
        for (int g = 0; g < 8; ++g) {
            float tm = fmaxf(s[0][g], s[1][g]);
#pragma unroll
            for (int off = 1; off < 16; off <<= 1) tm = fmaxf(tm, __shfl_xor(tm, off, 32));
            const float nm    = fmaxf(m_i[g], tm);
            const float alpha = __expf(m_i[g] - nm);
            m_i[g] = nm;
            const float p0 = __expf(s[0][g] - nm);
            const float p1 = __expf(s[1][g] - nm);
            s[0][g] = p0; s[1][g] = p1;
            float rs = p0 + p1;
#pragma unroll
            for (int off = 1; off < 16; off <<= 1) rs += __shfl_xor(rs, off, 32);
            l_i[g] = l_i[g] * alpha + rs;
#pragma unroll
            for (int nb = 0; nb < 4; ++nb) acc[nb][g] *= alpha;
        }

        // ---- P transpose via per-wave LDS slice (same-wave LDS is in-order) ----
#pragma unroll
        for (int nb = 0; nb < 2; ++nb)
#pragma unroll
            for (int g = 0; g < 8; ++g)
                pt[(g + 8 * half) * 32 + nb * 16 + l16] = (__bf16)s[nb][g];
        asm volatile("" ::: "memory");   // keep ds_load after ds_store
        const v16bf pa = ldA(pt + l16 * 32 + half * 16);
        asm volatile("" ::: "memory");

        // ---- O += P @ V ----
#pragma unroll
        for (int nb = 0; nb < 4; ++nb)
            acc[nb] = WMMA_BF16(pa, bv[nb], acc[nb]);

        // rotate K slab
        kb0[0] = kn0[0]; kb0[1] = kn0[1];
        kb1[0] = kn1[0]; kb1[1] = kn1[1];
        mb[0] = mbn[0]; mb[1] = mbn[1];
    }

    // ---- normalize, write bf16 attention output [B][N][C] ----
#pragma unroll
    for (int g = 0; g < 8; ++g) {
        const float inv = 1.0f / l_i[g];
        const int tok = q0 + g + 8 * half;
        __bf16* orow = out + ((size_t)bi * 2048 + tok) * 768 + h * 64;
#pragma unroll
        for (int nb = 0; nb < 4; ++nb)
            orow[nb * 16 + l16] = (__bf16)(acc[nb][g] * inv);
    }
}

// ---------------------------------------------------------------------------
// Kernel 3: output projection (bf16 x bf16 -> f32 + bias). Wave = 32x64 tile.
// ---------------------------------------------------------------------------
__global__ __launch_bounds__(256)
void proj_gemm_kernel(const __bf16* __restrict__ A, const __bf16* __restrict__ w,
                      const float* __restrict__ bias, float* __restrict__ out)
{
    const int lane = threadIdx.x & 31;
    const int wave = threadIdx.x >> 5;
    const int half = lane >> 4;
    const int l16  = lane & 15;
    const int m0   = blockIdx.y * 256 + wave * 32;
    const int n0   = blockIdx.x * 64;

    const __bf16* a0p = A + (size_t)(m0 + l16) * 768 + half * 16;
    const __bf16* a1p = a0p + (size_t)16 * 768;
    const __bf16* wp  = w + (size_t)(n0 + l16) * 768 + half * 16;

    v8f acc[2][4] = {};
    v16bf a[2], b[4];
    a[0] = ldA(a0p); a[1] = ldA(a1p);
#pragma unroll
    for (int nb = 0; nb < 4; ++nb) b[nb] = *(const v16bf*)(wp + (size_t)nb * 16 * 768);

#pragma unroll 2
    for (int k0 = 0; k0 < 768; k0 += 32) {
        const int kn = (k0 + 32 < 768) ? k0 + 32 : 0;
        v16bf an[2], bn[4];
        an[0] = ldA(a0p + kn); an[1] = ldA(a1p + kn);
#pragma unroll
        for (int nb = 0; nb < 4; ++nb) bn[nb] = *(const v16bf*)(wp + (size_t)nb * 16 * 768 + kn);
#pragma unroll
        for (int rg = 0; rg < 2; ++rg)
#pragma unroll
            for (int nb = 0; nb < 4; ++nb)
                acc[rg][nb] = WMMA_BF16(a[rg], b[nb], acc[rg][nb]);
        a[0] = an[0]; a[1] = an[1];
#pragma unroll
        for (int nb = 0; nb < 4; ++nb) b[nb] = bn[nb];
    }

#pragma unroll
    for (int nb = 0; nb < 4; ++nb) {
        const int col = n0 + nb * 16 + l16;
        const float bv = bias[col];
#pragma unroll
        for (int rg = 0; rg < 2; ++rg)
#pragma unroll
            for (int g = 0; g < 8; ++g) {
                const int m = m0 + rg * 16 + g + 8 * half;
                out[(size_t)m * 768 + col] = acc[rg][nb][g] + bv;
            }
    }
}

// ---------------------------------------------------------------------------
// Launcher
// ---------------------------------------------------------------------------
extern "C" void kernel_launch(void* const* d_in, const int* in_sizes, int n_in,
                              void* d_out, int out_size, void* d_ws, size_t ws_size,
                              hipStream_t stream)
{
    (void)in_sizes; (void)n_in; (void)out_size; (void)ws_size;
    const float* x      = (const float*)d_in[0];
    const float* mask   = (const float*)d_in[1];
    const float* qkv_w  = (const float*)d_in[2];
    const float* proj_w = (const float*)d_in[3];
    const float* proj_b = (const float*)d_in[4];
    float* out = (float*)d_out;

    // Workspace (bf16): Q 6M | K 6M | V^T 6M | AO 6M | xb 6M | qkv_wb 3.4M | proj_wb 1.2M
    char* ws = (char*)d_ws;
    __bf16* Q   = (__bf16*)(ws);
    __bf16* K   = (__bf16*)(ws + 6291456);
    __bf16* Vt  = (__bf16*)(ws + 12582912);
    __bf16* AO  = (__bf16*)(ws + 18874368);
    __bf16* xb  = (__bf16*)(ws + 25165824);
    __bf16* qwb = (__bf16*)(ws + 31457280);
    __bf16* pwb = (__bf16*)(ws + 34996224);

    // 0) one-shot f32 -> bf16 conversions (x, qkv_w, proj_w)
    cvt_bf16_kernel<<<dim3(3072), 256, 0, stream>>>(x, xb, 786432);       // 4096*768/4
    cvt_bf16_kernel<<<dim3(1728), 256, 0, stream>>>(qkv_w, qwb, 442368);  // 2304*768/4
    cvt_bf16_kernel<<<dim3(576),  256, 0, stream>>>(proj_w, pwb, 147456); // 768*768/4

    // 1) QKV projection: M=4096 (16 blocks of 256), N=2304 (36 blocks of 64)
    qkv_gemm_kernel<<<dim3(36, 16), 256, 0, stream>>>(xb, qwb, Q, K, Vt);

    // 2) Flash attention: 24 (b,h) * 128 query tiles, 4 waves/block
    flash_attn_kernel<<<dim3(768), 128, 0, stream>>>(Q, K, Vt, mask, AO);

    // 3) Output projection: N=768 (12 blocks of 64), M=4096 (16 blocks of 256)
    proj_gemm_kernel<<<dim3(12, 16), 256, 0, stream>>>(AO, pwb, proj_b, out);
}
